// AttentionBlock_79912161509545
// MI455X (gfx1250) — matrix operand order repaired and checked by reading, not compiled, not run on hardware
//
#include <hip/hip_runtime.h>

// ---------------- types ----------------
typedef __attribute__((ext_vector_type(16))) __bf16 v16bf;
typedef __attribute__((ext_vector_type(8)))  __bf16 v8bf;
typedef __attribute__((ext_vector_type(4)))  __bf16 v4bf;
typedef __attribute__((ext_vector_type(8)))  float  v8f;
typedef int v4i __attribute__((vector_size(16)));   // matches builtin param type
#define AS3 __attribute__((address_space(3)))

#define BATCH 4
#define SEQ   2048
#define DMODEL 1024
#define DQK   256
#define LDQK  512          // QK buffer row stride (Q cols 0..255, K cols 256..511)
#define NTOT  1536         // 256 Q + 256 K + 1024 V

// CDNA5 async global->LDS path (guarded; fallback = synchronous cooperative fill)
#if __has_builtin(__builtin_amdgcn_global_load_async_to_lds_b128) && \
    __has_builtin(__builtin_amdgcn_s_wait_asynccnt)
#define HAVE_ASYNC_LDS 1
#warning "CDNA5 async-to-LDS path ENABLED in this compilation pass"
#else
#define HAVE_ASYNC_LDS 0
#warning "CDNA5 async-to-LDS path DISABLED in this compilation pass"
#endif

// ---------------- helpers ----------------
__device__ __forceinline__ __bf16 f2bf(float f) {
  unsigned u = __float_as_uint(f);
  u += 0x7FFFu + ((u >> 16) & 1u);           // round-to-nearest-even
  return __builtin_bit_cast(__bf16, (unsigned short)(u >> 16));
}

__device__ __forceinline__ v8f zero8() {
  v8f z;
#pragma unroll
  for (int i = 0; i < 8; ++i) z[i] = 0.0f;
  return z;
}

__device__ __forceinline__ v16bf mk_frag(v8bf lo, v8bf hi) {
  return __builtin_shufflevector(lo, hi, 0,1,2,3,4,5,6,7,8,9,10,11,12,13,14,15);
}

// A-fragment (16xK row-major tile): lane row = l16; K = {h*8+0..7, h*8+16..23}
__device__ __forceinline__ v16bf load_a(const __bf16* rowptr, int hf) {
  const __bf16* p = rowptr + hf * 8;
  return mk_frag(*(const v8bf*)p, *(const v8bf*)(p + 16));
}
// B-fragment from B^T stored row-major (row = B column, contiguous K): K = {h*8+0..7, h*8+16..23}
__device__ __forceinline__ v16bf load_b(const __bf16* colptr, int hf) {
  const __bf16* p = colptr + hf * 8;
  return mk_frag(*(const v8bf*)p, *(const v8bf*)(p + 16));
}

__device__ __forceinline__ v8f wmma_bf16(v16bf a, v16bf b, v8f c) {
  return __builtin_amdgcn_wmma_f32_16x16x32_bf16(false, a, false, b, (short)0, c,
                                                 false, false);
}

__device__ __forceinline__ float rmax16(float v) {
#pragma unroll
  for (int m = 8; m >= 1; m >>= 1) v = fmaxf(v, __shfl_xor(v, m, 16));
  return v;
}
__device__ __forceinline__ float rsum16(float v) {
#pragma unroll
  for (int m = 8; m >= 1; m >>= 1) v += __shfl_xor(v, m, 16);
  return v;
}

// ---------------- kernel 1: x fp32 -> bf16 ----------------
__global__ __launch_bounds__(256) void conv_x_kernel(const float* __restrict__ x,
                                                     __bf16* __restrict__ xb, int n4) {
  int i = blockIdx.x * blockDim.x + threadIdx.x;
  if (i >= n4) return;
  float4 f = ((const float4*)x)[i];
  v4bf o;
  o[0] = f2bf(f.x); o[1] = f2bf(f.y); o[2] = f2bf(f.z); o[3] = f2bf(f.w);
  *(v4bf*)(xb + (size_t)i * 4) = o;
}

// ---------------- kernel 2: build Wt[1536][1024] = [Wq^T; Wk^T; Wv^T], bias cat ----------------
__global__ __launch_bounds__(256) void conv_w_kernel(
    const float* __restrict__ Wq, const float* __restrict__ Wk,
    const float* __restrict__ Wv, const float* __restrict__ bq,
    const float* __restrict__ bk, const float* __restrict__ bv,
    __bf16* __restrict__ Wt, float* __restrict__ bias) {
  int idx = blockIdx.x * 256 + threadIdx.x;
  if (idx < NTOT * DMODEL) {
    int n = idx >> 10, k = idx & 1023;
    float w = (n < 256)   ? Wq[k * DQK + n]
            : (n < 512)   ? Wk[k * DQK + (n - 256)]
                          : Wv[k * DMODEL + (n - 512)];
    Wt[idx] = f2bf(w);
  }
  if (idx < NTOT) {
    bias[idx] = (idx < 256) ? bq[idx] : (idx < 512) ? bk[idx - 256] : bv[idx - 512];
  }
}

// ---------------- kernel 3: projection GEMM  M=8192 K=1024 N=1536 ----------------
// 256 threads = 8 waves, wave tile 32(M)x64(N), workgroup tile 64(M)x256(N)
__global__ __launch_bounds__(256, 1) void proj_kernel(
    const __bf16* __restrict__ xb, const __bf16* __restrict__ Wt,
    const float* __restrict__ bias, __bf16* __restrict__ QK,
    __bf16* __restrict__ Vt) {
  const int lane = threadIdx.x & 31, wave = threadIdx.x >> 5;
  const int hf = lane >> 4, l16 = lane & 15;
  const int mwave = wave & 1, nwave = wave >> 1;
  const int mbase = blockIdx.x * 64 + mwave * 32;
  const int nbase = blockIdx.y * 256 + nwave * 64;

  v8f acc[2][4];
#pragma unroll
  for (int mt = 0; mt < 2; ++mt)
#pragma unroll
    for (int nt = 0; nt < 4; ++nt) acc[mt][nt] = zero8();

  const __bf16* Arow0 = xb + (size_t)(mbase + l16) * DMODEL;
  const __bf16* Arow1 = Arow0 + (size_t)16 * DMODEL;
  const __bf16* Brow  = Wt + (size_t)(nbase + l16) * DMODEL;

  for (int k0 = 0; k0 < DMODEL; k0 += 32) {
    v16bf a0 = load_a(Arow0 + k0, hf);
    v16bf a1 = load_a(Arow1 + k0, hf);
#pragma unroll
    for (int nt = 0; nt < 4; ++nt) {
      v16bf bfrag = load_b(Brow + (size_t)nt * 16 * DMODEL + k0, hf);
      acc[0][nt] = wmma_bf16(a0, bfrag, acc[0][nt]);
      acc[1][nt] = wmma_bf16(a1, bfrag, acc[1][nt]);
    }
  }

  if (nbase < 512) {
    // Q/K part: row-major [s][n], stride LDQK
#pragma unroll
    for (int nt = 0; nt < 4; ++nt) {
      int n = nbase + nt * 16 + l16;
      float bv = bias[n];
#pragma unroll
      for (int mt = 0; mt < 2; ++mt) {
#pragma unroll
        for (int v = 0; v < 8; ++v) {
          int m = mbase + mt * 16 + v + 8 * hf;
          QK[(size_t)m * LDQK + n] = f2bf(acc[mt][nt][v] + bv);
        }
      }
    }
  } else {
    // V part: store transposed Vt[b][d][s]; lane's 8 rows are contiguous s -> 16B store
#pragma unroll
    for (int nt = 0; nt < 4; ++nt) {
      int n = nbase + nt * 16 + l16;
      int d = n - 512;
      float bv = bias[n];
#pragma unroll
      for (int mt = 0; mt < 2; ++mt) {
        int m0 = mbase + mt * 16;
        int b  = m0 >> 11;
        int s0 = (m0 & (SEQ - 1)) + 8 * hf;
        v8bf pack;
#pragma unroll
        for (int v = 0; v < 8; ++v) pack[v] = f2bf(acc[mt][nt][v] + bv);
        *(v8bf*)(Vt + ((size_t)(b * DMODEL + d)) * SEQ + s0) = pack;
      }
    }
  }
}

// ---------------- kernel 4: flash attention + residual ----------------
// grid (S/16, B); 128 threads = 4 waves; wave w owns d-chunk [256w, 256w+256)
// K tile (32 keys x 256) staged once per workgroup into LDS (async on CDNA5),
// removing the 4x redundant global K traffic across the waves.
#define KROW 264   // padded LDS row stride: 528B = 33*16B (aligned, bank-spread)
__global__ __launch_bounds__(128, 1) void attn_kernel(
    const __bf16* __restrict__ QK, const __bf16* __restrict__ Vt,
    const float* __restrict__ x, float* __restrict__ out) {
  const int lane = threadIdx.x & 31, wave = threadIdx.x >> 5;
  const int hf = lane >> 4, l16 = lane & 15;
  const int q0 = blockIdx.x * 16;
  const int b  = blockIdx.y;
  const int dbase = wave * 256;

  __shared__ alignas(16) __bf16 Ks[32][KROW];    // staged K tile  (~16.9 KB)
  __shared__ alignas(16) __bf16 Ps[4][16][40];   // per-wave P transpose tile
  __bf16(*myP)[40] = Ps[wave];

  const __bf16* QKb = QK + (size_t)b * SEQ * LDQK;
  const __bf16* Qp  = QKb + (size_t)(q0 + l16) * LDQK;        // A rows (queries)
  const __bf16* Vb  = Vt + (size_t)b * DMODEL * SEQ;

  v8f acc[16];
#pragma unroll
  for (int t = 0; t < 16; ++t) acc[t] = zero8();
  float mi[8], li[8];
#pragma unroll
  for (int v = 0; v < 8; ++v) { mi[v] = -1e30f; li[v] = 0.0f; }

  for (int k0 = 0; k0 < SEQ; k0 += 32) {
    // ---- stage K tile [32 keys x 256] into LDS (once per workgroup) ----
    {
      const __bf16* Kg = QKb + (size_t)k0 * LDQK + DQK;   // K cols start at DQK
#pragma unroll
      for (int c = threadIdx.x; c < 1024; c += 128) {     // 1024 chunks of 16B
        int row = c >> 5, seg = c & 31;
        const __bf16* g = Kg + (size_t)row * LDQK + seg * 8;
        __bf16* l = &Ks[row][seg * 8];
#if HAVE_ASYNC_LDS
        __builtin_amdgcn_global_load_async_to_lds_b128(
            (v4i*)g, (AS3 v4i*)l, 0, 0);
#else
        *(v8bf*)l = *(const v8bf*)g;
#endif
      }
#if HAVE_ASYNC_LDS
      __builtin_amdgcn_s_wait_asynccnt(0);
#endif
      __syncthreads();
    }
    // ---- scores: Q[16x256] x K^T -> 16x32 (two 16x16 tiles), K frags from LDS ----
    v8f s0 = zero8(), s1 = zero8();
#pragma unroll
    for (int e = 0; e < 8; ++e) {
      const int off = e * 32;
      v16bf qa  = load_a(Qp + off, hf);
      v16bf kf0 = load_b(&Ks[l16][off], hf);
      v16bf kf1 = load_b(&Ks[16 + l16][off], hf);
      s0 = wmma_bf16(qa, kf0, s0);
      s1 = wmma_bf16(qa, kf1, s1);
    }
    // ---- online softmax (row stats per C-layout: VGPR v = row v + 8*hf) ----
    float sc[8];
#pragma unroll
    for (int v = 0; v < 8; ++v) {
      float tm = rmax16(fmaxf(s0[v], s1[v]));
      float mn = fmaxf(mi[v], tm);
      sc[v] = __expf(mi[v] - mn);
      float p0 = __expf(s0[v] - mn);
      float p1 = __expf(s1[v] - mn);
      li[v] = li[v] * sc[v] + rsum16(p0 + p1);
      mi[v] = mn;
      s0[v] = p0; s1[v] = p1;
    }
#pragma unroll
    for (int t = 0; t < 16; ++t)
#pragma unroll
      for (int v = 0; v < 8; ++v) acc[t][v] *= sc[v];
    // ---- C-layout -> A-layout through LDS (per-wave private tile) ----
#pragma unroll
    for (int v = 0; v < 8; ++v) {
      int row = v + 8 * hf;
      myP[row][l16]      = f2bf(s0[v]);
      myP[row][16 + l16] = f2bf(s1[v]);
    }
    v16bf pf = load_a(&myP[l16][0], hf);
    // ---- PV: P[16x32] x V[32 x 256-chunk], B-fragments from pre-transposed Vt ----
    const __bf16* Vp = Vb + (size_t)(dbase + l16) * SEQ + k0;
#pragma unroll
    for (int t = 0; t < 16; ++t) {
      v16bf vf = load_b(Vp + (size_t)t * 16 * SEQ, hf);
      acc[t] = wmma_bf16(pf, vf, acc[t]);
    }
    __syncthreads();   // all waves done reading Ks before next refill
  }

  // ---- epilogue: /l, + residual x ----
  float rli[8];
#pragma unroll
  for (int v = 0; v < 8; ++v) rli[v] = 1.0f / li[v];
  const float* xrow = x + (size_t)(b * SEQ + q0) * DMODEL;
  float* orow = out + (size_t)(b * SEQ + q0) * DMODEL;
#pragma unroll
  for (int t = 0; t < 16; ++t) {
    int d = dbase + t * 16 + l16;
#pragma unroll
    for (int v = 0; v < 8; ++v) {
      size_t idx = (size_t)(v + 8 * hf) * DMODEL + d;
      orow[idx] = acc[t][v] * rli[v] + xrow[idx];
    }
  }
}

// ---------------- launcher ----------------
extern "C" void kernel_launch(void* const* d_in, const int* in_sizes, int n_in,
                              void* d_out, int out_size, void* d_ws, size_t ws_size,
                              hipStream_t stream) {
  const float* x  = (const float*)d_in[0];
  const float* Wq = (const float*)d_in[1];
  const float* bq = (const float*)d_in[2];
  const float* Wk = (const float*)d_in[3];
  const float* bk = (const float*)d_in[4];
  const float* Wv = (const float*)d_in[5];
  const float* bv = (const float*)d_in[6];
  float* outp = (float*)d_out;

  const size_t M = (size_t)BATCH * SEQ;                // 8192
  char* w = (char*)d_ws;
  __bf16* xb = (__bf16*)w;  w += M * DMODEL * 2;                 // 16.78 MB
  __bf16* Wt = (__bf16*)w;  w += (size_t)NTOT * DMODEL * 2;      //  3.15 MB
  float* bias = (float*)w;  w += (size_t)NTOT * 4 + 256;         //  6.4 KB
  __bf16* QKb = (__bf16*)w; w += M * LDQK * 2;                   //  8.39 MB
  __bf16* Vtb = (__bf16*)w; w += (size_t)BATCH * DMODEL * SEQ * 2; // 16.78 MB
  if ((size_t)(w - (char*)d_ws) > ws_size) return;     // insufficient scratch

  // 1) convert x to bf16
  int n4 = (int)(M * DMODEL / 4);
  conv_x_kernel<<<(n4 + 255) / 256, 256, 0, stream>>>(x, xb, n4);
  // 2) transpose/concat weights to bf16 + bias cat
  conv_w_kernel<<<(NTOT * DMODEL + 255) / 256, 256, 0, stream>>>(Wq, Wk, Wv, bq, bk,
                                                                 bv, Wt, bias);
  // 3) QKV projection GEMM (WMMA)
  proj_kernel<<<dim3((int)(M / 64), NTOT / 256), 256, 0, stream>>>(xb, Wt, bias, QKb,
                                                                   Vtb);
  // 4) flash attention + residual (WMMA)
  attn_kernel<<<dim3(SEQ / 16, BATCH), 128, 0, stream>>>(QKb, Vtb, x, outp);
}
